// VectorQuantizer_15771119911129
// MI455X (gfx1250) — matrix operand order repaired and checked by reading, not compiled, not run on hardware
//
#include <hip/hip_runtime.h>
#include <math.h>

// ---------------------------------------------------------------------------
// VQ-VAE vector quantizer for MI455X (gfx1250, wave32, WMMA + TDM).
//   x:          [131072, 256] f32 tokens
//   embeddings: [1024, 256]   f32 codebook
//   out:        quantized [131072,256] f32  ++  perplexity (1 f32)
// Distance argmin via  d = ||e||^2 - 2*x.e  computed with split-bf16 GEMM
// (a_hi*b_hi + a_lo*b_hi + a_hi*b_lo) on v_wmma_f32_16x16x32_bf16.
// Codebook panels are streamed into LDS with the Tensor Data Mover
// (tensor_load_to_lds, TENSORcnt), double-buffered against the WMMA loop.
// ---------------------------------------------------------------------------

typedef __attribute__((ext_vector_type(16))) __bf16 v16bf;
typedef __attribute__((ext_vector_type(8)))  __bf16 v8bf;
typedef __attribute__((ext_vector_type(8)))  float  v8f;
typedef unsigned int u32x4 __attribute__((ext_vector_type(4)));
typedef int          i32x8 __attribute__((ext_vector_type(8)));
typedef int          i32x4 __attribute__((ext_vector_type(4)));

#define VQ_NTOK 131072
#define VQ_D    256
#define VQ_K    1024
#define VQ_NCHUNK 64            // 64 chunks of 16 codes
#define VQ_CHUNK_ELTS 8192      // (hi 16x256 + lo 16x256) bf16 per chunk

// Split a float into hi/lo bf16 parts and place at vector element J.
#define VQ_CVT(HV, LV, F, J)                                   \
  {                                                            \
    __bf16 _h = (__bf16)(F);                                   \
    (HV)[J] = _h;                                              \
    (LV)[J] = (__bf16)((F) - (float)_h);                       \
  }

// ---------------------------------------------------------------------------
// TDM: copy one 16 KB chunk panel (global, contiguous) -> LDS.
// D# per CDNA5 ISA: 1-row tile, data_size=8B,
// tensor_dim0 = tile_dim0 = 2048 units, tensor_dim1 = tile_dim1 = 1.
// This toolchain's builtin takes 6 args (g0, g1, g2, g3, extra, cpol).
// ---------------------------------------------------------------------------
__device__ __forceinline__ void vq_tdm_load_16k(const __bf16* gsrc,
                                                unsigned int lds_byte) {
  unsigned long long ga = (unsigned long long)(uintptr_t)gsrc;
  u32x4 g0;
  g0[0] = 1u;                                   // count=1 (valid user D#)
  g0[1] = lds_byte;                             // lds_addr (bytes)
  g0[2] = (unsigned int)ga;                     // global_addr[31:0]
  g0[3] = (unsigned int)(ga >> 32) | 0x80000000u; // global_addr[56:32] | type=2
  i32x8 g1;
  g1[0] = (3 << 16);                            // data_size = 3 (8-byte units)
  g1[1] = (int)(2048u << 16);                   // tensor_dim0[15:0] = 2048
  g1[2] = (1 << 16);                            // tensor_dim0 hi = 0; tensor_dim1 = 1
  g1[3] = (int)(2048u << 16);                   // tile_dim0 = 2048
  g1[4] = 1;                                    // tile_dim1 = 1, tile_dim2 = 0
  g1[5] = 2048;                                 // tensor_dim0_stride = 2048
  g1[6] = 0;
  g1[7] = 0;
  i32x4 z4 = {0, 0, 0, 0};                      // groups 2/3 unused (<=2D)
  i32x8 z8 = {0, 0, 0, 0, 0, 0, 0, 0};          // extra operand (unused)
  __builtin_amdgcn_tensor_load_to_lds(g0, g1, z4, z4, z8, 0);
}

// ---------------------------------------------------------------------------
// Prep: per code c (one wave per block): ||e_c||^2, split-bf16 codebook packed
// per 16-code chunk as [chunk][hi 16x256 | lo 16x256], zero histogram bin c.
// ---------------------------------------------------------------------------
__global__ __launch_bounds__(32) void vq_prep(
    const float* __restrict__ emb,
    __bf16* __restrict__ epack,
    float* __restrict__ normE, float* __restrict__ counts) {
  const int c    = blockIdx.x;        // 0..1023
  const int lane = threadIdx.x;       // 0..31, 8 dims per lane
  const float* src = emb + (size_t)c * VQ_D + lane * 8;

  float s = 0.0f;
  v8bf hv, lv;
  float f;
  f = src[0]; s += f * f; VQ_CVT(hv, lv, f, 0);
  f = src[1]; s += f * f; VQ_CVT(hv, lv, f, 1);
  f = src[2]; s += f * f; VQ_CVT(hv, lv, f, 2);
  f = src[3]; s += f * f; VQ_CVT(hv, lv, f, 3);
  f = src[4]; s += f * f; VQ_CVT(hv, lv, f, 4);
  f = src[5]; s += f * f; VQ_CVT(hv, lv, f, 5);
  f = src[6]; s += f * f; VQ_CVT(hv, lv, f, 6);
  f = src[7]; s += f * f; VQ_CVT(hv, lv, f, 7);

  const int chunk = c >> 4;           // 16-code chunk id
  const int ci    = c & 15;           // code within chunk
  __bf16* dh = epack + (size_t)chunk * VQ_CHUNK_ELTS + ci * VQ_D + lane * 8;
  __bf16* dl = dh + 4096;             // lo panel follows hi panel
  *(v8bf*)dh = hv;                    // 16B aligned
  *(v8bf*)dl = lv;

  #pragma unroll
  for (int off = 16; off >= 1; off >>= 1)
    s += __shfl_xor(s, off, 32);

  if (lane == 0) {
    normE[c]  = s;
    counts[c] = 0.0f;   // re-zero histogram every call (harness requirement)
  }
}

// ---------------------------------------------------------------------------
// Main: 8 waves/block, each wave owns 16 tokens. 64 chunks of 16 codes,
// TDM-streamed double-buffered B panels in LDS; 3 WMMAs per K-step.
// Fused: argmin reduce, fp32 gather of winning code rows, histogram atomics.
// ---------------------------------------------------------------------------
__global__ __launch_bounds__(256) void vq_main(
    const float* __restrict__ x, const float* __restrict__ emb,
    const __bf16* __restrict__ epack,
    const float* __restrict__ normE, float* __restrict__ counts,
    float* __restrict__ outq) {
  // [buffer][hi 16x256 | lo 16x256] bf16 = 2 x 16 KB
  __shared__ __align__(1024) __bf16 panel[2][VQ_CHUNK_ELTS];

  const int tid   = threadIdx.x;
  const int wave  = tid >> 5;          // 0..7
  const int lane  = tid & 31;
  const int half  = lane >> 4;         // 0: rows 0-7 / K-lo ; 1: rows 8-15 / K-hi
  const int l16   = lane & 15;
  const int tokBase = (blockIdx.x * 8 + wave) * 16;
  const bool issuer = (wave == 0);     // one wave drives the TDM

  // Kick the TDM for chunk 0 before touching x, so the 16 KB L2->LDS
  // transfer overlaps the A-tile preload below.
  if (issuer)
    vq_tdm_load_16k(epack, (unsigned int)(uintptr_t)&panel[0][0]);

  // --- Preload A tile (16 tokens x 256 dims) as split-bf16 fragments -------
  // ISA 16-bit A 16x32 layout: lanes 0-15 -> M=lane, K = {0..7, 16..23};
  // lanes 16-31 -> M=lane-16, K = {8..15, 24..31} (per 32-wide K step).
  v16bf aH[8], aL[8];
  const float* xrow = x + (size_t)(tokBase + l16) * VQ_D;
  #pragma unroll
  for (int ks = 0; ks < 8; ++ks) {
    const int base = ks * 32 + half * 8;
    float4 f0 = *(const float4*)(xrow + base);
    float4 f1 = *(const float4*)(xrow + base + 4);
    float4 f2 = *(const float4*)(xrow + base + 16);
    float4 f3 = *(const float4*)(xrow + base + 20);
    v16bf ah, al;
    VQ_CVT(ah, al, f0.x, 0);  VQ_CVT(ah, al, f0.y, 1);
    VQ_CVT(ah, al, f0.z, 2);  VQ_CVT(ah, al, f0.w, 3);
    VQ_CVT(ah, al, f1.x, 4);  VQ_CVT(ah, al, f1.y, 5);
    VQ_CVT(ah, al, f1.z, 6);  VQ_CVT(ah, al, f1.w, 7);
    VQ_CVT(ah, al, f2.x, 8);  VQ_CVT(ah, al, f2.y, 9);
    VQ_CVT(ah, al, f2.z, 10); VQ_CVT(ah, al, f2.w, 11);
    VQ_CVT(ah, al, f3.x, 12); VQ_CVT(ah, al, f3.y, 13);
    VQ_CVT(ah, al, f3.z, 14); VQ_CVT(ah, al, f3.w, 15);
    aH[ks] = ah;
    aL[ks] = al;
  }

  float minv[8];
  int   mini[8];
  #pragma unroll
  for (int r = 0; r < 8; ++r) { minv[r] = 3.4e38f; mini[r] = 0; }

  if (issuer) __builtin_amdgcn_s_wait_tensorcnt(0);
  __syncthreads();

  // --- Sweep the codebook: double-buffered TDM + WMMA ----------------------
  for (int k = 0; k < VQ_NCHUNK; ++k) {
    const int cur = k & 1;

    // Issue TDM for the next chunk into the other buffer (its previous
    // consumers finished at the barrier that ended iteration k-1).
    if (issuer && (k + 1 < VQ_NCHUNK))
      vq_tdm_load_16k(epack + (size_t)(k + 1) * VQ_CHUNK_ELTS,
                      (unsigned int)(uintptr_t)&panel[cur ^ 1][0]);

    const __bf16* ph = &panel[cur][0];
    const __bf16* pl = &panel[cur][4096];
    const int codeId = k * 16 + l16;    // code this lane scores
    const float ne   = normE[codeId];

    v8f c = {};
    #pragma unroll
    for (int ks = 0; ks < 8; ++ks) {
      // ISA 16-bit B 32x16 layout: lanes 0-15 -> N=lane, K={0..7,16..23};
      // lanes 16-31 -> N=lane-16, K={8..15,24..31} (per 32-wide K step).
      const int boff = l16 * VQ_D + ks * 32 + half * 8;
      v8bf bh0 = *(const v8bf*)(ph + boff);
      v8bf bh1 = *(const v8bf*)(ph + boff + 16);
      v8bf bl0 = *(const v8bf*)(pl + boff);
      v8bf bl1 = *(const v8bf*)(pl + boff + 16);
      v16bf bh = __builtin_shufflevector(bh0, bh1, 0,1,2,3,4,5,6,7,8,9,10,11,12,13,14,15);
      v16bf bl = __builtin_shufflevector(bl0, bl1, 0,1,2,3,4,5,6,7,8,9,10,11,12,13,14,15);
      c = __builtin_amdgcn_wmma_f32_16x16x32_bf16(false, aH[ks], false, bh,
                                                  (short)0, c, false, false);
      c = __builtin_amdgcn_wmma_f32_16x16x32_bf16(false, aL[ks], false, bh,
                                                  (short)0, c, false, false);
      c = __builtin_amdgcn_wmma_f32_16x16x32_bf16(false, aH[ks], false, bl,
                                                  (short)0, c, false, false);
    }

    // C layout: VGPR r -> row (r + 8*half), col = l16.  dist = ||e||^2 - 2 x.e
    #pragma unroll
    for (int r = 0; r < 8; ++r) {
      float d = ne - 2.0f * c[r];
      if (d < minv[r]) { minv[r] = d; mini[r] = codeId; }
    }

    // Next-chunk panel must be resident before anyone computes on it.
    if (issuer) __builtin_amdgcn_s_wait_tensorcnt(0);
    __syncthreads();
  }

  // --- Argmin across the 16 lanes of each half (xor masks stay in-half) ----
  #pragma unroll
  for (int r = 0; r < 8; ++r) {
    #pragma unroll
    for (int off = 8; off >= 1; off >>= 1) {
      float ov = __shfl_xor(minv[r], off, 32);
      int   oi = __shfl_xor(mini[r], off, 32);
      if (ov < minv[r] || (ov == minv[r] && oi < mini[r])) {
        minv[r] = ov; mini[r] = oi;
      }
    }
  }

  // --- Fused gather (fp32 codebook -> out) + histogram ---------------------
  #pragma unroll
  for (int r = 0; r < 8; ++r) {
    const int row = tokBase + half * 8 + r;
    const int idx = mini[r];
    const float4* src = (const float4*)(emb  + (size_t)idx * VQ_D + l16 * 16);
    float4*       dst = (float4*)(outq + (size_t)row * VQ_D + l16 * 16);
    dst[0] = src[0]; dst[1] = src[1]; dst[2] = src[2]; dst[3] = src[3];
    if (l16 == r) atomicAdd(counts + idx, 1.0f);
  }
}

// ---------------------------------------------------------------------------
// Perplexity: exp(-sum p log(p + 1e-10)), p = counts / N.
// ---------------------------------------------------------------------------
__global__ __launch_bounds__(256) void vq_perplexity(
    const float* __restrict__ counts, float* __restrict__ outp) {
  __shared__ float red[256];
  float s = 0.0f;
  for (int b = threadIdx.x; b < VQ_K; b += 256) {
    float p = counts[b] * (1.0f / (float)VQ_NTOK);
    s += p * logf(p + 1e-10f);
  }
  red[threadIdx.x] = s;
  __syncthreads();
  for (int st = 128; st > 0; st >>= 1) {
    if (threadIdx.x < st) red[threadIdx.x] += red[threadIdx.x + st];
    __syncthreads();
  }
  if (threadIdx.x == 0) *outp = expf(-red[0]);
}

// ---------------------------------------------------------------------------
extern "C" void kernel_launch(void* const* d_in, const int* in_sizes, int n_in,
                              void* d_out, int out_size, void* d_ws, size_t ws_size,
                              hipStream_t stream) {
  const float* x   = (const float*)d_in[0];   // [131072, 256]
  const float* emb = (const float*)d_in[1];   // [1024, 256]
  float* out = (float*)d_out;                 // quantized ++ perplexity

  // Workspace layout (~1.03 MB):
  //   [0,4K)       normE   (1024 f32)
  //   [4K,8K)      counts  (1024 f32)
  //   [8K,1032K)   epack   64 chunks x [hi 16x256 | lo 16x256] bf16
  char* ws = (char*)d_ws;
  float*  normE  = (float*)ws;
  float*  counts = (float*)(ws + 4096);
  __bf16* epack  = (__bf16*)(ws + 8192);

  vq_prep<<<VQ_K, 32, 0, stream>>>(emb, epack, normE, counts);
  vq_main<<<VQ_NTOK / 128, 256, 0, stream>>>(x, emb, epack, normE, counts, out);
  vq_perplexity<<<1, 256, 0, stream>>>(counts, out + (size_t)(out_size - 1));
}
